// Qwen3OmniCodePredictorAttention_49478023250555
// MI455X (gfx1250) — matrix operand order repaired and checked by reading, not compiled, not run on hardware
//
#include <hip/hip_runtime.h>
#include <hip/hip_bf16.h>
#include <cstdint>

// ---------------- problem constants ----------------
#define B_   4
#define S_   2048
#define H_   1024
#define NH_  16
#define NKV_ 4
#define HD_  64
#define NREP_ 4
#define SCALE_ 0.125f   // 1/sqrt(64)

typedef __bf16 bf16;
typedef __attribute__((ext_vector_type(16))) __bf16 v16bf;
typedef __attribute__((ext_vector_type(8)))  float  v8f;
typedef int v4i __attribute__((vector_size(16)));

struct alignas(16) U128 { unsigned int x[4]; };
union FragB { v16bf v; U128 q[2]; };
struct alignas(8) BF4 { bf16 h[4]; };

static __device__ __forceinline__ v8f v8zero() {
  v8f z = {0.f,0.f,0.f,0.f,0.f,0.f,0.f,0.f};
  return z;
}

// ---------------- async global->LDS staging (CDNA5 async-DMA path) ----------------
#if __has_builtin(__builtin_amdgcn_global_load_async_to_lds_b128)
#define HAVE_ASYNC_LDS 1
#else
#define HAVE_ASYNC_LDS 0
#endif

static __device__ __forceinline__ void stage16(const bf16* g, bf16* l) {
#if HAVE_ASYNC_LDS
  // builtin signature: (v4i addrspace(1)*, v4i addrspace(3)*, imm offset, imm cpol)
  // LDS address = low 32 bits of the generic address (LDS aperture truncation)
  __builtin_amdgcn_global_load_async_to_lds_b128(
      (__attribute__((address_space(1))) v4i*)(uintptr_t)g,
      (__attribute__((address_space(3))) v4i*)(uint32_t)(uintptr_t)l, 0, 0);
#else
  *(U128*)l = *(const U128*)g;
#endif
}
static __device__ __forceinline__ void stage_wait() {
#if HAVE_ASYNC_LDS
  asm volatile("s_wait_asynccnt 0" ::: "memory");
#endif
}

// ---------------- cast fp32 -> bf16 (row-major copy) ----------------
__global__ void cast_f32_bf16(const float* __restrict__ in, bf16* __restrict__ out, int n4) {
  int i = blockIdx.x * blockDim.x + threadIdx.x;
  if (i >= n4) return;
  float4 f = ((const float4*)in)[i];
  BF4 o;
  o.h[0] = (bf16)f.x; o.h[1] = (bf16)f.y; o.h[2] = (bf16)f.z; o.h[3] = (bf16)f.w;
  ((BF4*)out)[i] = o;
}

// ---------------- cast + transpose: in [K][N] fp32 -> out [N][K] bf16 ----------------
__global__ void cast_transpose_f32_bf16(const float* __restrict__ in, bf16* __restrict__ out,
                                        int K, int N) {
  int idx = blockIdx.x * blockDim.x + threadIdx.x;
  if (idx >= N * K) return;
  int n = idx / K;
  int k = idx % K;
  out[idx] = (bf16)in[(size_t)k * N + n];
}

// ---------------- WMMA GEMM: C[M,N] = A[M,K] * Bt[N,K]^T ----------------
// A [M][K] bf16, Bt [N][K] bf16 (pre-transposed weights), C fp32.
// M,N multiples of 128; K multiple of 32. Double-buffered async LDS staging.
#define BM 128
#define BN 128
#define BK 32

__global__ __launch_bounds__(256) void gemm_bf16(const bf16* __restrict__ A,
                                                 const bf16* __restrict__ Bt,
                                                 float* __restrict__ C,
                                                 int M, int N, int K) {
  __shared__ __align__(16) bf16 As[2][BM * BK];   // [m][k]
  __shared__ __align__(16) bf16 Bs[2][BN * BK];   // [n][k]
  const int t    = threadIdx.x;
  const int lane = t & 31;
  const int wid  = t >> 5;
  const int wm   = wid & 3;     // wave M position (4)
  const int wn   = wid >> 2;    // wave N position (2)
  const int bm   = blockIdx.y * BM;
  const int bn   = blockIdx.x * BN;
  const int lh   = lane & 15;
  const int lg   = lane >> 4;

  v8f acc[2][4];
#pragma unroll
  for (int i = 0; i < 2; ++i)
#pragma unroll
    for (int j = 0; j < 4; ++j) acc[i][j] = v8zero();

  const int srow = t >> 2;            // 0..63 (two rows per thread: +0, +64)
  const int skg  = (t & 3) * 8;       // k group of 8 halfs (16B)

  auto stage = [&](int kt, int buf) {
#pragma unroll
    for (int r2 = 0; r2 < 2; ++r2) {
      int row = srow + r2 * 64;
      stage16(A  + (size_t)(bm + row) * K + kt + skg, &As[buf][row * BK + skg]);
      stage16(Bt + (size_t)(bn + row) * K + kt + skg, &Bs[buf][row * BK + skg]);
    }
  };

  const int nk = K / BK;
  stage(0, 0);

  for (int it = 0; it < nk; ++it) {
    const int buf = it & 1;
    stage_wait();          // own async deposits done
    __syncthreads();       // everyone's deposits visible; prev buffer free
    if (it + 1 < nk) stage((it + 1) * BK, buf ^ 1);
    if (it + 2 < nk) {     // warm far caches two tiles ahead
      __builtin_prefetch(A  + (size_t)(bm + srow) * K + (it + 2) * BK + skg, 0, 3);
      __builtin_prefetch(Bt + (size_t)(bn + srow) * K + (it + 2) * BK + skg, 0, 3);
    }

    // ---- fragments from current buffer ----
    FragB af[2], bfr[4];
#pragma unroll
    for (int i = 0; i < 2; ++i) {
      const bf16* p = &As[buf][(wm * 32 + i * 16 + lh) * BK + lg * 8];
      af[i].q[0] = *(const U128*)p;
      af[i].q[1] = *(const U128*)(p + 16);
    }
#pragma unroll
    for (int j = 0; j < 4; ++j) {
      const bf16* p = &Bs[buf][(wn * 64 + j * 16 + lh) * BK + lg * 8];
      bfr[j].q[0] = *(const U128*)p;
      bfr[j].q[1] = *(const U128*)(p + 16);
    }
#pragma unroll
    for (int i = 0; i < 2; ++i)
#pragma unroll
      for (int j = 0; j < 4; ++j)
        acc[i][j] = __builtin_amdgcn_wmma_f32_16x16x32_bf16(
            false, af[i].v, false, bfr[j].v, (short)0, acc[i][j], false, false);
    // frag reads are consumed by WMMA (s_wait_dscnt) before next top barrier,
    // so the single barrier per iteration is sufficient with double buffering.
  }

  // ---- write back (C/D layout: M = v + 8*laneHalf, N = lane&15) ----
#pragma unroll
  for (int i = 0; i < 2; ++i)
#pragma unroll
    for (int j = 0; j < 4; ++j)
#pragma unroll
      for (int v = 0; v < 8; ++v) {
        int row = bm + wm * 32 + i * 16 + v + 8 * lg;
        int col = bn + wn * 64 + j * 16 + lh;
        C[(size_t)row * N + col] = acc[i][j][v];
      }
}

// ---------------- fused RMSNorm + RoPE, pack bf16 [b][h][s][d] ----------------
__global__ void qk_norm_rope(const float* __restrict__ in, const float* __restrict__ cosb,
                             const float* __restrict__ sinb, const float* __restrict__ w,
                             bf16* __restrict__ out, int nh) {
  int idx = blockIdx.x * blockDim.x + threadIdx.x;
  if (idx >= B_ * S_ * nh) return;
  int h = idx % nh;
  int r = idx / nh;         // token row b*S+s
  int s = r % S_;
  int b = r / S_;
  const float* x = in + (size_t)r * (nh * HD_) + h * HD_;
  float xv[HD_];
  float ss = 0.f;
#pragma unroll
  for (int d = 0; d < HD_; ++d) { xv[d] = x[d]; ss += xv[d] * xv[d]; }
  float inv = rsqrtf(ss * (1.0f / HD_) + 1e-6f);
  bf16* o = out + (((size_t)b * nh + h) * S_ + s) * HD_;
#pragma unroll
  for (int i = 0; i < HD_ / 2; ++i) {
    float c  = cosb[s * HD_ + 2 * i];
    float sn = sinb[s * HD_ + 2 * i];
    float x0 = xv[2 * i] * inv * w[2 * i];
    float x1 = xv[2 * i + 1] * inv * w[2 * i + 1];
    o[2 * i]     = (bf16)(x0 * c - x1 * sn);
    o[2 * i + 1] = (bf16)(x1 * c + x0 * sn);
  }
}

// ---------------- V: cast + transpose to [b][kvh][d][s] ----------------
__global__ void v_transpose(const float* __restrict__ Vf, bf16* __restrict__ Vt) {
  int idx = blockIdx.x * blockDim.x + threadIdx.x;
  if (idx >= B_ * S_ * NKV_) return;
  int h = idx % NKV_;
  int r = idx / NKV_;
  int s = r % S_;
  int b = r / S_;
  const float* x = Vf + (size_t)r * (NKV_ * HD_) + h * HD_;
  bf16* o = Vt + (((size_t)b * NKV_ + h) * HD_) * (size_t)S_ + s;
#pragma unroll
  for (int d = 0; d < HD_; ++d) o[(size_t)d * S_] = (bf16)x[d];
}

// ---------------- flash attention: 1 wave = 16 q rows ----------------
__global__ __launch_bounds__(128) void attn_flash(const bf16* __restrict__ Q,
                                                  const bf16* __restrict__ Kc,
                                                  const bf16* __restrict__ Vt,
                                                  bf16* __restrict__ attn) {
  __shared__ __align__(16) bf16 Plds[4][16 * 32];
  const int lane = threadIdx.x & 31;
  const int wid  = threadIdx.x >> 5;
  const int gt   = blockIdx.x * 4 + wid;   // 0..8191 q-tiles
  const int qt   = gt & 127;
  const int bh   = gt >> 7;
  const int h    = bh % NH_;
  const int b    = bh / NH_;
  const int kvh  = h >> 2;                 // GQA: NREP=4
  const int qbase = qt * 16;
  const int lh = lane & 15;
  const int lg = lane >> 4;

  // Q fragments (A layout, 16x32 per d-half)
  FragB qf[2];
  const bf16* Qrow = Q + (((size_t)b * NH_ + h) * S_ + qbase + lh) * HD_;
#pragma unroll
  for (int c = 0; c < 2; ++c) {
    const bf16* p = Qrow + c * 32 + lg * 8;
    qf[c].q[0] = *(const U128*)p;
    qf[c].q[1] = *(const U128*)(p + 16);
  }
  const bf16* Kbase = Kc + (((size_t)b * NKV_ + kvh) * S_) * HD_;
  const bf16* Vbase = Vt + (((size_t)b * NKV_ + kvh) * HD_) * (size_t)S_;

  v8f o[4];
  float m[8], l[8];
#pragma unroll
  for (int c = 0; c < 4; ++c) o[c] = v8zero();
#pragma unroll
  for (int v = 0; v < 8; ++v) { m[v] = -1e30f; l[v] = 0.f; }

  bf16* P = Plds[wid];
  const int nkc = (qbase + 47) >> 5;   // key chunks of 32
  for (int kc = 0; kc < nkc; ++kc) {
    const int kstart = kc * 32;
    // ---- scores: two 16x16 tiles over keys ----
    v8f sc[2];
#pragma unroll
    for (int t2 = 0; t2 < 2; ++t2) {
      sc[t2] = v8zero();
#pragma unroll
      for (int dh = 0; dh < 2; ++dh) {
        FragB kf;
        const bf16* p = Kbase + (size_t)(kstart + t2 * 16 + lh) * HD_ + dh * 32 + lg * 8;
        kf.q[0] = *(const U128*)p;
        kf.q[1] = *(const U128*)(p + 16);
        sc[t2] = __builtin_amdgcn_wmma_f32_16x16x32_bf16(
            false, qf[dh].v, false, kf.v, (short)0, sc[t2], false, false);
      }
    }
    // ---- scale + causal mask ----
    const bool need_mask = (kstart + 31) > qbase;
#pragma unroll
    for (int t2 = 0; t2 < 2; ++t2)
#pragma unroll
      for (int v = 0; v < 8; ++v) {
        float s = sc[t2][v] * SCALE_;
        if (need_mask) {
          int kg = kstart + t2 * 16 + lh;
          int qg = qbase + v + 8 * lg;
          if (kg > qg) s += -1e9f;
        }
        sc[t2][v] = s;
      }
    // ---- online softmax (row reductions across the 16-lane half) ----
    float mn[8], alpha[8];
#pragma unroll
    for (int v = 0; v < 8; ++v) {
      float mx = fmaxf(sc[0][v], sc[1][v]);
#pragma unroll
      for (int off = 1; off < 16; off <<= 1) mx = fmaxf(mx, __shfl_xor(mx, off, 32));
      mn[v] = fmaxf(m[v], mx);
      alpha[v] = __expf(m[v] - mn[v]);
      m[v] = mn[v];
    }
#pragma unroll
    for (int v = 0; v < 8; ++v) {
      float p0 = __expf(sc[0][v] - mn[v]);
      float p1 = __expf(sc[1][v] - mn[v]);
      sc[0][v] = p0; sc[1][v] = p1;
      float s2 = p0 + p1;
#pragma unroll
      for (int off = 1; off < 16; off <<= 1) s2 += __shfl_xor(s2, off, 32);
      l[v] = l[v] * alpha[v] + s2;
    }
#pragma unroll
    for (int c = 0; c < 4; ++c)
#pragma unroll
      for (int v = 0; v < 8; ++v) o[c][v] *= alpha[v];
    // ---- P: C layout -> LDS row-major [16][32] -> A layout ----
#pragma unroll
    for (int t2 = 0; t2 < 2; ++t2)
#pragma unroll
      for (int v = 0; v < 8; ++v)
        P[(v + 8 * lg) * 32 + t2 * 16 + lh] = (bf16)sc[t2][v];
    asm volatile("s_wait_dscnt 0" ::: "memory");   // wave-local LDS WAR/RAW fence
    FragB pf;
    {
      const bf16* pp = P + lh * 32 + lg * 8;
      pf.q[0] = *(const U128*)pp;
      pf.q[1] = *(const U128*)(pp + 16);
    }
    // ---- O += P * V ----
#pragma unroll
    for (int c = 0; c < 4; ++c) {
      FragB vf;
      const bf16* vp = Vbase + (size_t)(c * 16 + lh) * S_ + kstart + lg * 8;
      vf.q[0] = *(const U128*)vp;
      vf.q[1] = *(const U128*)(vp + 16);
      o[c] = __builtin_amdgcn_wmma_f32_16x16x32_bf16(
          false, pf.v, false, vf.v, (short)0, o[c], false, false);
    }
    asm volatile("s_wait_dscnt 0" ::: "memory");
  }
  // ---- normalize + write attn [b][s][h*64+d] as bf16 ----
#pragma unroll
  for (int c = 0; c < 4; ++c)
#pragma unroll
    for (int v = 0; v < 8; ++v) {
      int sq = qbase + v + 8 * lg;
      float val = o[c][v] / l[v];
      attn[((size_t)b * S_ + sq) * (NH_ * HD_) + h * HD_ + c * 16 + lh] = (bf16)val;
    }
}

// ---------------- host orchestration ----------------
extern "C" void kernel_launch(void* const* d_in, const int* in_sizes, int n_in,
                              void* d_out, int out_size, void* d_ws, size_t ws_size,
                              hipStream_t stream) {
  (void)in_sizes; (void)n_in; (void)out_size; (void)ws_size;
  const float* hs   = (const float*)d_in[0];
  /* d_in[1] causal_mask: applied analytically */
  const float* cosb = (const float*)d_in[2];
  const float* sinb = (const float*)d_in[3];
  const float* Wq   = (const float*)d_in[4];
  const float* Wk   = (const float*)d_in[5];
  const float* Wv   = (const float*)d_in[6];
  const float* Wo   = (const float*)d_in[7];
  const float* qw   = (const float*)d_in[8];
  const float* kw   = (const float*)d_in[9];
  float* out = (float*)d_out;

  char* ws = (char*)d_ws;
  size_t off = 0;
  auto alloc = [&](size_t bytes) -> void* {
    void* p = ws + off;
    off += (bytes + 255) & ~(size_t)255;
    return p;
  };
  const size_t T = (size_t)B_ * S_;            // 8192 tokens
  bf16* hs_bf  = (bf16*)alloc(T * H_ * 2);
  bf16* WqT_bf = (bf16*)alloc((size_t)(NH_ * HD_) * H_ * 2);   // [N][K]
  bf16* WkT_bf = (bf16*)alloc((size_t)(NKV_ * HD_) * H_ * 2);
  bf16* WvT_bf = (bf16*)alloc((size_t)(NKV_ * HD_) * H_ * 2);
  bf16* WoT_bf = (bf16*)alloc((size_t)H_ * (NH_ * HD_) * 2);
  float* Qf    = (float*)alloc(T * (NH_ * HD_) * 4);
  float* Kf    = (float*)alloc(T * (NKV_ * HD_) * 4);
  float* Vf    = (float*)alloc(T * (NKV_ * HD_) * 4);
  bf16* Qbf    = (bf16*)alloc(T * (NH_ * HD_) * 2);
  bf16* Kbf    = (bf16*)alloc(T * (NKV_ * HD_) * 2);
  bf16* Vtbf   = (bf16*)alloc(T * (NKV_ * HD_) * 2);
  bf16* attnb  = (bf16*)Qf;   // alias: Qf retired before attention runs

  const int TPB = 256;
  {  // hidden: straight cast
    int n4 = (int)(T * H_ / 4);
    cast_f32_bf16<<<(n4 + TPB - 1) / TPB, TPB, 0, stream>>>(hs, hs_bf, n4);
  }
  auto castT = [&](const float* src, bf16* dst, int K, int N) {
    int n = N * K;
    cast_transpose_f32_bf16<<<(n + TPB - 1) / TPB, TPB, 0, stream>>>(src, dst, K, N);
  };
  castT(Wq, WqT_bf, H_, NH_ * HD_);
  castT(Wk, WkT_bf, H_, NKV_ * HD_);
  castT(Wv, WvT_bf, H_, NKV_ * HD_);
  castT(Wo, WoT_bf, NH_ * HD_, H_);

  // QKV projections
  gemm_bf16<<<dim3((NH_ * HD_) / BN, (int)T / BM), 256, 0, stream>>>(
      hs_bf, WqT_bf, Qf, (int)T, NH_ * HD_, H_);
  gemm_bf16<<<dim3((NKV_ * HD_) / BN, (int)T / BM), 256, 0, stream>>>(
      hs_bf, WkT_bf, Kf, (int)T, NKV_ * HD_, H_);
  gemm_bf16<<<dim3((NKV_ * HD_) / BN, (int)T / BM), 256, 0, stream>>>(
      hs_bf, WvT_bf, Vf, (int)T, NKV_ * HD_, H_);

  // norm + rope + pack
  {
    int nq = (int)(T * NH_);
    qk_norm_rope<<<(nq + TPB - 1) / TPB, TPB, 0, stream>>>(Qf, cosb, sinb, qw, Qbf, NH_);
    int nk = (int)(T * NKV_);
    qk_norm_rope<<<(nk + TPB - 1) / TPB, TPB, 0, stream>>>(Kf, cosb, sinb, kw, Kbf, NKV_);
    v_transpose<<<(nk + TPB - 1) / TPB, TPB, 0, stream>>>(Vf, Vtbf);
  }

  // flash attention: 8192 q-tiles, 4 waves per block
  attn_flash<<<(B_ * NH_ * (S_ / 16)) / 4, 128, 0, stream>>>(Qbf, Kbf, Vtbf, attnb);

  // output projection -> fp32 d_out
  gemm_bf16<<<dim3(H_ / BN, (int)T / BM), 256, 0, stream>>>(
      attnb, WoT_bf, out, (int)T, H_, NH_ * HD_);
}
